// AttentionLayer_54245436948579
// MI455X (gfx1250) — matrix operand order repaired and checked by reading, not compiled, not run on hardware
//
#include <hip/hip_runtime.h>
#include <hip/hip_bf16.h>

#define BB 64
#define TT 2048
#define DD 256

typedef __attribute__((ext_vector_type(16))) __bf16 bf16x16;
typedef __attribute__((ext_vector_type(8)))  __bf16 bf16x8;
typedef __attribute__((ext_vector_type(8)))  float  f32x8;

// Fast tanh: prefer the gfx1250 hardware v_tanh_f32; otherwise a branch-free
// rational form 1 - 2/(e^{2x}+1) built on v_exp_f32 + v_rcp_f32.
// (exp overflow -> +inf -> result 1.0; exp underflow -> 0 -> result -1.0)
__device__ __forceinline__ float fast_tanh(float x)
{
#if defined(__has_builtin)
#if __has_builtin(__builtin_amdgcn_tanhf)
    return __builtin_amdgcn_tanhf(x);
#else
    const float e = __builtin_amdgcn_exp2f(x * 2.88539008177792681472f); // 2*log2(e)
    return 1.0f - 2.0f * __builtin_amdgcn_rcpf(e + 1.0f);
#endif
#else
    const float e = __builtin_amdgcn_exp2f(x * 2.88539008177792681472f);
    return 1.0f - 2.0f * __builtin_amdgcn_rcpf(e + 1.0f);
#endif
}

// ---------------------------------------------------------------------------
// Kernel 1: u = tanh(A * W^T + bias); scores = u . v      (A = 64-row strip)
// Grid: (B*T)/64 = 2048 blocks, 256 threads (8 waves).
// Each wave: one 16-row tile x 8 column tiles, K=256 via 8x wmma 16x16x32 bf16
// ---------------------------------------------------------------------------
__global__ __launch_bounds__(256) void gemm_tanh_scores(
    const float* __restrict__ hs, const float* __restrict__ W,
    const float* __restrict__ bias, const float* __restrict__ av,
    float* __restrict__ scores)
{
    __shared__ __bf16 Wb[DD * DD];   // 128 KB, layout [n][k] (B matrix, col n)
    __shared__ __bf16 As[64 * DD];   //  32 KB, layout [r][k]
    __shared__ float  biasS[DD];
    __shared__ float  vS[DD];
    __shared__ float  spart[8][16];

    const int tid = threadIdx.x;

    // W is [e][d] row-major == exactly B[n][k]: straight copy + f32->bf16
    const float4* W4 = (const float4*)W;
    #pragma unroll 4
    for (int i = tid; i < DD * DD / 4; i += 256) {
        float4 w = W4[i];
        Wb[i * 4 + 0] = (__bf16)w.x;  Wb[i * 4 + 1] = (__bf16)w.y;
        Wb[i * 4 + 2] = (__bf16)w.z;  Wb[i * 4 + 3] = (__bf16)w.w;
    }
    // 64-row contiguous A strip for this block
    const float4* A4 = (const float4*)(hs + (size_t)blockIdx.x * 64 * DD);
    #pragma unroll 4
    for (int i = tid; i < 64 * DD / 4; i += 256) {
        float4 a = A4[i];
        As[i * 4 + 0] = (__bf16)a.x;  As[i * 4 + 1] = (__bf16)a.y;
        As[i * 4 + 2] = (__bf16)a.z;  As[i * 4 + 3] = (__bf16)a.w;
    }
    if (tid < DD) { biasS[tid] = bias[tid]; vS[tid] = av[tid]; }
    __syncthreads();

    const int w    = tid >> 5;        // wave id 0..7
    const int lane = tid & 31;
    const int rt   = w & 3;           // row tile 0..3
    const int ch   = w >> 2;          // column half 0..1
    const int l16  = lane & 15;
    const int hf   = lane >> 4;       // lane half
    const int arow = rt * 16 + l16;

    // Hoist the 8 A fragments (ISA 16-bit A 16x32 layout):
    //  lanes 0-15 : K = k0+{0..7}, k0+{16..23};  lanes 16-31: +8 on each run
    bf16x16 afrag[8];
    #pragma unroll
    for (int k = 0; k < 8; ++k) {
        const __bf16* base = &As[arow * DD + k * 32 + hf * 8];
        ((bf16x8*)&afrag[k])[0] = *(const bf16x8*)(base);
        ((bf16x8*)&afrag[k])[1] = *(const bf16x8*)(base + 16);
    }

    float racc[8] = {0.f, 0.f, 0.f, 0.f, 0.f, 0.f, 0.f, 0.f};

    // Unroll by 2 (not 8): enough to double-buffer B fragments against WMMA
    // latency, while staying under 256 VGPRs (full unroll forced s_set_vgpr_msb
    // and >320 VGPRs/wave, capping occupancy at 3 waves/SIMD).
    #pragma unroll 2
    for (int c8 = 0; c8 < 8; ++c8) {
        const int ct = ch * 8 + c8;
        const int n  = ct * 16 + l16;           // output column for this lane
        // B fragment: lanes 0-15 hold K=k0+{0..7},k0+{16..23} of column n, lanes 16-31 +8
        const __bf16* brow = &Wb[n * DD + hf * 8];
        f32x8 acc = {};
        #pragma unroll
        for (int k = 0; k < 8; ++k) {
            bf16x16 bfrag;
            ((bf16x8*)&bfrag)[0] = *(const bf16x8*)(brow + k * 32);
            ((bf16x8*)&bfrag)[1] = *(const bf16x8*)(brow + k * 32 + 16);
            acc = __builtin_amdgcn_wmma_f32_16x16x32_bf16(
                false, afrag[k], false, bfrag, (short)0, acc, false, false);
        }
        const float bn = biasS[n];
        const float vn = vS[n];
        // C layout: VGPR i = row (rt*16 + i) for lanes 0-15, (rt*16+8+i) for 16-31
        #pragma unroll
        for (int i = 0; i < 8; ++i)
            racc[i] += fast_tanh(acc[i] + bn) * vn;
    }

    // Reduce across the 16 columns held in each lane half
    #pragma unroll
    for (int i = 0; i < 8; ++i) {
        float x = racc[i];
        x += __shfl_xor(x, 1, 32);
        x += __shfl_xor(x, 2, 32);
        x += __shfl_xor(x, 4, 32);
        x += __shfl_xor(x, 8, 32);
        if (lane == 0)  spart[w][i]     = x;   // row rt*16 + i
        if (lane == 16) spart[w][8 + i] = x;   // row rt*16 + 8 + i
    }
    __syncthreads();

    if (tid < 64) {
        const int r = tid;
        float s = spart[r >> 4][r & 15] + spart[(r >> 4) + 4][r & 15];
        scores[(size_t)blockIdx.x * 64 + r] = s;   // flat index m = b*T + t
    }
}

// ---------------------------------------------------------------------------
// Kernel 2: softmax over axis 0 (batch), in place.  One thread per t column.
// ---------------------------------------------------------------------------
__global__ __launch_bounds__(256) void softmax_axis0(float* __restrict__ sc)
{
    const int t = blockIdx.x * 256 + threadIdx.x;
    float v[BB];
    float m = -__builtin_inff();
    #pragma unroll
    for (int b = 0; b < BB; ++b) {
        v[b] = sc[(size_t)b * TT + t];
        m = fmaxf(m, v[b]);
    }
    float s = 0.f;
    #pragma unroll
    for (int b = 0; b < BB; ++b) {
        v[b] = __expf(v[b] - m);
        s += v[b];
    }
    const float inv = 1.0f / s;
    #pragma unroll
    for (int b = 0; b < BB; ++b)
        sc[(size_t)b * TT + t] = v[b] * inv;
}

// ---------------------------------------------------------------------------
// Kernel 3: partial s[b,d] over 128-step t chunks (deterministic, no atomics)
// Grid: (16 chunks, 64 batches), 256 threads (one per d).
// ---------------------------------------------------------------------------
__global__ __launch_bounds__(256) void wsum_partial(
    const float* __restrict__ hs, const float* __restrict__ a,
    float* __restrict__ partial)
{
    const int b = blockIdx.y, c = blockIdx.x, d = threadIdx.x;
    const float* ab = a  + (size_t)b * TT + c * 128;
    const float* hb = hs + ((size_t)b * TT + (size_t)c * 128) * DD + d;
    float acc = 0.f;
    #pragma unroll 8
    for (int t = 0; t < 128; ++t)
        acc = fmaf(ab[t], hb[(size_t)t * DD], acc);
    partial[((size_t)b * 16 + c) * DD + d] = acc;
}

// ---------------------------------------------------------------------------
// Kernel 4: fixed-order reduction of the 16 chunks -> d_out [B, D]
// ---------------------------------------------------------------------------
__global__ __launch_bounds__(256) void wsum_reduce(
    const float* __restrict__ partial, float* __restrict__ out)
{
    const int b = blockIdx.x, d = threadIdx.x;
    float acc = 0.f;
    #pragma unroll
    for (int c = 0; c < 16; ++c)
        acc += partial[((size_t)b * 16 + c) * DD + d];
    out[(size_t)b * DD + d] = acc;
}

extern "C" void kernel_launch(void* const* d_in, const int* in_sizes, int n_in,
                              void* d_out, int out_size, void* d_ws, size_t ws_size,
                              hipStream_t stream)
{
    const float* hs   = (const float*)d_in[0];   // [64, 2048, 256]
    const float* W    = (const float*)d_in[1];   // [256, 256]
    const float* bias = (const float*)d_in[2];   // [256]
    const float* av   = (const float*)d_in[3];   // [256]
    float* out = (float*)d_out;                  // [64, 256]

    float* scores  = (float*)d_ws;               // [64*2048] = 512 KB (reused as 'a')
    float* partial = scores + (size_t)BB * TT;   // [64*16*256] = 1 MB

    gemm_tanh_scores<<<dim3((BB * TT) / 64), 256, 0, stream>>>(hs, W, bias, av, scores);
    softmax_axis0  <<<dim3(TT / 256),        256, 0, stream>>>(scores);
    wsum_partial   <<<dim3(16, BB),          256, 0, stream>>>(hs, scores, partial);
    wsum_reduce    <<<dim3(BB),              256, 0, stream>>>(partial, out);
}
